// CrossMatch_8211977470153
// MI455X (gfx1250) — matrix operand order repaired and checked by reading, not compiled, not run on hardware
//
#include <hip/hip_runtime.h>
#include <hip/hip_bf16.h>
#include <math.h>

// ---------------------------------------------------------------------------
// CDNA5 (gfx1250) 2-layer cross-attention encoder.
// GEMMs: v_wmma_f32_16x16x32_bf16 (wave32, f32 accumulate).
// gemm_proj uses the Tensor Data Mover (tensor_load_to_lds + s_wait_tensorcnt)
// with double-buffered LDS A tiles.
// ---------------------------------------------------------------------------

typedef __bf16 bf16;
typedef __attribute__((ext_vector_type(16))) __bf16 v16bf;
typedef __attribute__((ext_vector_type(8)))  __bf16 v8bf;
typedef __attribute__((ext_vector_type(8)))  float  v8f;
typedef __attribute__((ext_vector_type(4)))  unsigned int v4u;
typedef __attribute__((ext_vector_type(4)))  int v4i;
typedef __attribute__((ext_vector_type(8)))  int v8i;

union bfrag { v16bf v; v8bf h[2]; };

#define NB   16        // batch
#define LL   4800      // tokens (80*60)
#define CC   256       // d_model
#define NHD  8         // heads
#define HH_  80
#define WW_  60
#define KV_SPLIT 10    // S split for KV reduction (4800/10 = 480 = 15*32)

__device__ inline bf16 f2bf(float f) {
  unsigned u = __builtin_bit_cast(unsigned, f);
  unsigned r = (u + 0x7FFFu + ((u >> 16) & 1u)) >> 16;
  return __builtin_bit_cast(bf16, (unsigned short)r);
}
__device__ inline float bf2f(bf16 b) {
  unsigned u = ((unsigned)__builtin_bit_cast(unsigned short, b)) << 16;
  return __builtin_bit_cast(float, u);
}
__device__ inline v8f wmma_bf16(v16bf a, v16bf b, v8f c) {
  return __builtin_amdgcn_wmma_f32_16x16x32_bf16(false, a, false, b,
                                                 (short)0, c, false, false);
}

// ---------------------------------------------------------------------------
// TDM: issue tensor_load_to_lds of a (tile_h x tile_w) bf16 tile, row stride
// stride_elems (elements), into LDS offset lds_off. Wave-level op (EXEC
// ignored); caller must restrict to one wave and fence with tensorcnt+barrier.
// D# field packing per CDNA5 ISA ch.8 (group0: count/lds/global/type,
// group1: data_size, tensor dims, tile dims, dim0 stride).
// ---------------------------------------------------------------------------
__device__ __forceinline__ void tdm_load_tile(const void* gaddr, unsigned lds_off,
                                              unsigned tile_w, unsigned tile_h,
                                              unsigned stride_elems) {
  unsigned long long ga = (unsigned long long)(uintptr_t)gaddr;
  v4u g0;
  g0.x = 1u;                                            // count=1, user mode
  g0.y = lds_off;                                       // lds_addr
  g0.z = (unsigned)ga;                                  // global_addr[31:0]
  g0.w = (unsigned)((ga >> 32) & 0x01FFFFFFu) | (2u << 30); // addr[56:32]|type=2
  v8i g1;
  g1[0] = (int)(1u << 16);                              // data_size = 2 bytes
  g1[1] = (int)((tile_w & 0xFFFFu) << 16);              // tensor_dim0[15:0]
  g1[2] = (int)((tile_w >> 16) | ((tile_h & 0xFFFFu) << 16)); // td0 hi | td1 lo
  g1[3] = (int)((tile_h >> 16) | (tile_w << 16));       // td1 hi | tile_dim0
  g1[4] = (int)tile_h;                                  // tile_dim1 (tile_dim2=0)
  g1[5] = (int)stride_elems;                            // dim0_stride[31:0]
  g1[6] = 0;                                            // dim0_stride hi | d1s lo
  g1[7] = 0;
  v4i z4 = {0, 0, 0, 0};
#if __clang_major__ >= 23
  v8i z8 = {0, 0, 0, 0, 0, 0, 0, 0};
  __builtin_amdgcn_tensor_load_to_lds(g0, g1, z4, z4, z8, 0);
#else
  __builtin_amdgcn_tensor_load_to_lds(g0, g1, z4, z4, 0);
#endif
}

// ---------------------------------------------------------------------------
// Weight pre-pack: fp32 [K,256] row-major -> bf16 WMMA-B fragment-major.
// ---------------------------------------------------------------------------
__global__ void pack_w_kernel(const float* __restrict__ src,
                              bf16* __restrict__ dst, int Kdim) {
  int t = blockIdx.x * 256 + threadIdx.x;
  if (t >= Kdim * 256) return;
  int s      = t & 15;
  int lane   = (t >> 4) & 31;
  int tile   = t >> 9;
  int tile_n = tile & 15;
  int tile_k = tile >> 4;
  int n  = tile_n * 16 + (lane & 15);
  int kb = (lane < 16) ? 0 : 8;
  int k  = tile_k * 32 + kb + (s < 8 ? s : s + 8);
  dst[t] = f2bf(src[k * 256 + n]);
}

// ---------------------------------------------------------------------------
// Positional encoding add + relayout NCHW -> [N*L, C].
// Quirk preserved: fac = (-log(1e4)/256)//2 == -1, so div_j = exp(-2j).
// ---------------------------------------------------------------------------
__global__ void pe_pack_kernel(const float* __restrict__ feat,
                               float* __restrict__ Xf32,
                               bf16* __restrict__ xbf, int ldx, int write_f32) {
  int t = blockIdx.x * 256 + threadIdx.x;
  if (t >= NB * CC * HH_ * WW_) return;
  int w = t % WW_;
  int h = (t / WW_) % HH_;
  int c = (t / (WW_ * HH_)) % CC;
  int n = t / (WW_ * HH_ * CC);
  int j   = c >> 2;
  int rem = c & 3;
  float divj = __expf(-2.0f * (float)j);
  float pos  = (rem < 2) ? (float)(w + 1) : (float)(h + 1);
  float ang  = pos * divj;
  float pe   = (rem & 1) ? __cosf(ang) : __sinf(ang);
  float v = feat[t] + pe;
  size_t row = (size_t)n * LL + (size_t)h * WW_ + w;
  xbf[row * ldx + c] = f2bf(v);
  if (write_f32) Xf32[row * CC + c] = v;
}

// ---------------------------------------------------------------------------
// Main GEMM: C[M,256-slice] = A[M,Kdim](bf16,lda) x Bpacked. BM=64, BN=128,
// 8 waves (4M x 2N), each wave 16x64 (4 wmma accumulators).
// A tiles arrive via TDM (tensor_load_to_lds), double-buffered in LDS;
// wave 0 issues tile tk+1 while all waves compute tile tk.
// mode 0: relu   -> out[row*256+col]
// mode 1: elu+1  -> Q  [ (n*8+h)*L + l ][d]
// mode 2: elu+1  -> Kt [ (n*8+h)*32 + d ][s]
// mode 3: *1/S   -> V  [ (n*8+h)*L + s ][d]
// ---------------------------------------------------------------------------
__global__ __launch_bounds__(256) void gemm_proj_kernel(
    const bf16* __restrict__ A, int lda, const bf16* __restrict__ Bp,
    int Kdim, int mode, bf16* __restrict__ out) {
  __shared__ __align__(16) bf16 Atile[2][64 * 32];
  const int lane  = threadIdx.x & 31;
  const int wv    = threadIdx.x >> 5;
  const int waveM = wv & 3;
  const int waveN = wv >> 2;
  const int m0 = blockIdx.y * 64;
  const int n0 = blockIdx.x * 128;
  v8f acc[4] = {};
  const int tiles_k = Kdim >> 5;
  const unsigned ldsoff[2] = {(unsigned)(uintptr_t)&Atile[0][0],
                              (unsigned)(uintptr_t)&Atile[1][0]};
  const bf16* Arow = A + (size_t)m0 * lda;
  // per-wave packed-B base: tiles are 512 bf16; 16 n-tiles per k-tile.
  const bf16* Bw = Bp + ((size_t)(((n0 >> 4) + waveN * 4) * 32 + lane)) * 16;

  if (wv == 0) tdm_load_tile(Arow, ldsoff[0], 32, 64, (unsigned)lda);

  for (int tk = 0; tk < tiles_k; ++tk) {
    if (wv == 0) {
      if (tk + 1 < tiles_k) {
        tdm_load_tile(Arow + (tk + 1) * 32, ldsoff[(tk + 1) & 1], 32, 64,
                      (unsigned)lda);
        __builtin_amdgcn_s_wait_tensorcnt(1);   // tile tk has landed
      } else {
        __builtin_amdgcn_s_wait_tensorcnt(0);
      }
    }
    __syncthreads();                            // tile tk visible to all waves
    const bf16* At = &Atile[tk & 1][0];
    bfrag af;
    const int am = waveM * 16 + (lane & 15);
    const int kb = (lane < 16) ? 0 : 8;
    af.h[0] = *(const v8bf*)&At[am * 32 + kb];
    af.h[1] = *(const v8bf*)&At[am * 32 + kb + 16];
    bfrag bm[4];
    const bf16* bt = Bw + (size_t)tk * 8192;    // 16 tiles * 512 bf16
#pragma unroll
    for (int a = 0; a < 4; ++a) {
      bm[a].h[0] = *(const v8bf*)(bt + a * 512);
      bm[a].h[1] = *(const v8bf*)(bt + a * 512 + 8);
    }
#pragma unroll
    for (int a = 0; a < 4; ++a)
      acc[a] = wmma_bf16(af.v, bm[a].v, acc[a]);
    __syncthreads();                            // reads done before buffer reuse
  }
#pragma unroll
  for (int a = 0; a < 4; ++a) {
    const int col = n0 + waveN * 64 + a * 16 + (lane & 15);
#pragma unroll
    for (int r = 0; r < 8; ++r) {
      const int row = m0 + waveM * 16 + r + ((lane < 16) ? 0 : 8);
      float v = acc[a][r];
      if (mode == 0) {
        out[(size_t)row * CC + col] = f2bf(v > 0.f ? v : 0.f);
      } else {
        const int n = row / LL, l = row % LL;
        const int hh = col >> 5, d = col & 31;
        if (mode == 1) {
          v = (v > 0.f) ? v + 1.f : __expf(v);
          out[(((size_t)(n * NHD + hh) * LL + l) << 5) + d] = f2bf(v);
        } else if (mode == 2) {
          v = (v > 0.f) ? v + 1.f : __expf(v);
          out[((size_t)(n * NHD + hh) * 32 + d) * LL + l] = f2bf(v);
        } else {
          out[(((size_t)(n * NHD + hh) * LL + l) << 5) + d] =
              f2bf(v * (1.0f / (float)LL));
        }
      }
    }
  }
}

// ---------------------------------------------------------------------------
// GEMM + fused LayerNorm. BM=32, BN=256 (full row), 8 waves (2M x 4N).
// mode 0 (ln1): Concat[row][256+c] = LN(acc)      (Wm output)
// mode 1 (ln2): x += LN(acc); Xf32, Concat[row][c] updated (W2 output)
// ---------------------------------------------------------------------------
__global__ __launch_bounds__(256) void gemm_ln_kernel(
    const bf16* __restrict__ A, const bf16* __restrict__ Bp,
    const float* __restrict__ g, const float* __restrict__ b, int mode,
    bf16* __restrict__ concat, float* __restrict__ Xf32) {
  __shared__ __align__(16) bf16 Atile[32 * 32];
  __shared__ float Cls[32][264];
  __shared__ float rsum[8][32], rsq[8][32];
  __shared__ float smu[32], srs[32];
  const int lane  = threadIdx.x & 31;
  const int wv    = threadIdx.x >> 5;
  const int waveN = wv & 3;
  const int waveM = wv >> 2;
  const int m0 = blockIdx.x * 32;
  v8f acc[4] = {};
  const bf16* Bw = Bp + ((size_t)((waveN * 4) * 32 + lane)) * 16;
  for (int tk = 0; tk < 8; ++tk) {
    const int k0 = tk << 5;
    if (threadIdx.x < 128) {
      const int ldrow = threadIdx.x >> 2, ldseg = threadIdx.x & 3;
      *(uint4*)&Atile[ldrow * 32 + ldseg * 8] =
          *(const uint4*)&A[(size_t)(m0 + ldrow) * CC + k0 + ldseg * 8];
    }
    __syncthreads();
    bfrag af;
    const int am = waveM * 16 + (lane & 15);
    const int kb = (lane < 16) ? 0 : 8;
    af.h[0] = *(const v8bf*)&Atile[am * 32 + kb];
    af.h[1] = *(const v8bf*)&Atile[am * 32 + kb + 16];
    bfrag bm[4];
    const bf16* bt = Bw + (size_t)tk * 8192;
#pragma unroll
    for (int a = 0; a < 4; ++a) {
      bm[a].h[0] = *(const v8bf*)(bt + a * 512);
      bm[a].h[1] = *(const v8bf*)(bt + a * 512 + 8);
    }
#pragma unroll
    for (int a = 0; a < 4; ++a)
      acc[a] = wmma_bf16(af.v, bm[a].v, acc[a]);
    __syncthreads();
  }
#pragma unroll
  for (int a = 0; a < 4; ++a) {
    const int col = waveN * 64 + a * 16 + (lane & 15);
#pragma unroll
    for (int r = 0; r < 8; ++r) {
      const int row = waveM * 16 + r + ((lane < 16) ? 0 : 8);
      Cls[row][col] = acc[a][r];
    }
  }
  __syncthreads();
  {
    const int row = threadIdx.x & 31, ch = threadIdx.x >> 5;
    float s = 0.f, q = 0.f;
    for (int c = ch * 32; c < ch * 32 + 32; ++c) {
      float x = Cls[row][c];
      s += x; q += x * x;
    }
    rsum[ch][row] = s; rsq[ch][row] = q;
  }
  __syncthreads();
  if (threadIdx.x < 32) {
    float s = 0.f, q = 0.f;
    for (int ch = 0; ch < 8; ++ch) { s += rsum[ch][threadIdx.x]; q += rsq[ch][threadIdx.x]; }
    const float mu  = s * (1.f / 256.f);
    const float var = q * (1.f / 256.f) - mu * mu;
    smu[threadIdx.x] = mu;
    srs[threadIdx.x] = rsqrtf(var + 1e-5f);
  }
  __syncthreads();
  {
    const int row = threadIdx.x >> 3;
    const int c0  = (threadIdx.x & 7) * 32;
    const size_t grow = (size_t)(m0 + row);
    const float mu = smu[row], rs = srs[row];
    for (int c = c0; c < c0 + 32; ++c) {
      float val = (Cls[row][c] - mu) * rs * g[c] + b[c];
      if (mode == 0) {
        concat[grow * 512 + 256 + c] = f2bf(val);
      } else {
        const size_t ix = grow * CC + c;
        float x = Xf32[ix] + val;
        Xf32[ix] = x;
        concat[grow * 512 + c] = f2bf(x);
      }
    }
  }
}

// ---------------------------------------------------------------------------
// KV = K^T (V/S) per (n,h): 32x32 over S via WMMA; S split across blocks,
// accumulated with global f32 atomics. Also accumulates Ksum[d].
// ---------------------------------------------------------------------------
__global__ __launch_bounds__(128) void kv_kernel(
    const bf16* __restrict__ Kt, const bf16* __restrict__ V,
    float* __restrict__ KV, float* __restrict__ Ksum) {
  __shared__ __align__(16) bf16 Klds[32 * 32];   // [d][s]
  __shared__ __align__(16) bf16 Vlds[32 * 32];   // [v][s] (transposed)
  const int nh = blockIdx.y;
  const int s0base = blockIdx.x * (LL / KV_SPLIT);
  const int lane = threadIdx.x & 31;
  const int wv = threadIdx.x >> 5;
  const int tm = wv & 1, tn = wv >> 1;
  v8f acc = {};
  float ksum_part = 0.f;
  const int ldrow = threadIdx.x >> 2, ldseg = threadIdx.x & 3;
  for (int it = 0; it < (LL / KV_SPLIT) / 32; ++it) {
    const int s0 = s0base + it * 32;
    *(uint4*)&Klds[ldrow * 32 + ldseg * 8] =
        *(const uint4*)&Kt[((size_t)nh * 32 + ldrow) * LL + s0 + ldseg * 8];
    {
      v8bf tmp = *(const v8bf*)&V[((size_t)nh * LL + s0 + ldrow) * 32 + ldseg * 8];
#pragma unroll
      for (int j = 0; j < 8; ++j) Vlds[(ldseg * 8 + j) * 32 + ldrow] = tmp[j];
    }
    __syncthreads();
    {
      const int d = threadIdx.x & 31, p = threadIdx.x >> 5;
#pragma unroll
      for (int j = 0; j < 8; ++j) ksum_part += bf2f(Klds[d * 32 + p * 8 + j]);
    }
    bfrag af, bm;
    const int am = tm * 16 + (lane & 15);
    const int kb = (lane < 16) ? 0 : 8;
    af.h[0] = *(const v8bf*)&Klds[am * 32 + kb];
    af.h[1] = *(const v8bf*)&Klds[am * 32 + kb + 16];
    const int vn  = tn * 16 + (lane & 15);
    const int kb2 = (lane < 16) ? 0 : 8;
    bm.h[0] = *(const v8bf*)&Vlds[vn * 32 + kb2];
    bm.h[1] = *(const v8bf*)&Vlds[vn * 32 + kb2 + 16];
    acc = wmma_bf16(af.v, bm.v, acc);
    __syncthreads();
  }
  atomicAdd(&Ksum[nh * 32 + (threadIdx.x & 31)], ksum_part);
#pragma unroll
  for (int r = 0; r < 8; ++r) {
    const int d  = tm * 16 + r + ((lane < 16) ? 0 : 8);
    const int vn = tn * 16 + (lane & 15);
    atomicAdd(&KV[(size_t)nh * 1024 + d * 32 + vn], acc[r]);
  }
}

// Z[n,l,h] = 1 / (Q . Ksum + eps)
__global__ void z_kernel(const bf16* __restrict__ Q,
                         const float* __restrict__ Ksum,
                         float* __restrict__ Z) {
  int t = blockIdx.x * 256 + threadIdx.x;
  if (t >= NB * LL * NHD) return;
  const int hh = t & 7;
  const int l  = (t >> 3) % LL;
  const int n  = t / (LL * NHD);
  const bf16*  q  = Q + ((size_t)(n * NHD + hh) * LL + l) * 32;
  const float* ks = Ksum + (n * NHD + hh) * 32;
  float d = 0.f;
#pragma unroll
  for (int i = 0; i < 32; ++i) d += bf2f(q[i]) * ks[i];
  Z[t] = 1.0f / (d + 1e-6f);
}

// out[n,l,h*32+v] = (Q[n,h,l,:] @ KV[n,h]) * Z * S   (M=L, N=32, K=32 WMMA)
__global__ __launch_bounds__(128) void attn_kernel(
    const bf16* __restrict__ Q, const float* __restrict__ KV,
    const float* __restrict__ Z, bf16* __restrict__ out) {
  __shared__ __align__(16) bf16 kvb[32 * 32];  // [v][k]
  const int nh = blockIdx.y;
  const int m0 = blockIdx.x * 64;
  for (int e = threadIdx.x; e < 1024; e += 128) {
    const int k = e >> 5, vv = e & 31;
    kvb[vv * 32 + k] = f2bf(KV[(size_t)nh * 1024 + e]);
  }
  __syncthreads();
  const int lane = threadIdx.x & 31;
  const int wv = threadIdx.x >> 5;
  const int am = m0 + wv * 16 + (lane & 15);
  const int kb = (lane < 16) ? 0 : 8;
  const bf16* qr = Q + ((size_t)nh * LL + am) * 32;
  bfrag af;
  af.h[0] = *(const v8bf*)(qr + kb);
  af.h[1] = *(const v8bf*)(qr + kb + 16);
  v8f acc[2] = {};
#pragma unroll
  for (int a = 0; a < 2; ++a) {
    const int vn  = a * 16 + (lane & 15);
    const int kb2 = (lane < 16) ? 0 : 8;
    bfrag bm;
    bm.h[0] = *(const v8bf*)&kvb[vn * 32 + kb2];
    bm.h[1] = *(const v8bf*)&kvb[vn * 32 + kb2 + 16];
    acc[a] = wmma_bf16(af.v, bm.v, acc[a]);
  }
  const int n = nh >> 3, hh = nh & 7;
#pragma unroll
  for (int a = 0; a < 2; ++a) {
#pragma unroll
    for (int r = 0; r < 8; ++r) {
      const int row = m0 + wv * 16 + r + ((lane < 16) ? 0 : 8);
      const int col = hh * 32 + a * 16 + (lane & 15);
      const float z = Z[((size_t)n * LL + row) * NHD + hh];
      out[((size_t)n * LL + row) * CC + col] = f2bf(acc[a][r] * z * (float)LL);
    }
  }
}

__global__ void zero_kernel(float* p, int count) {
  int t = blockIdx.x * 256 + threadIdx.x;
  if (t < count) p[t] = 0.f;
}

// [N,L,C] fp32 -> [N,C,H,W] fp32
__global__ void final_kernel(const float* __restrict__ X, float* __restrict__ out) {
  int t = blockIdx.x * 256 + threadIdx.x;
  if (t >= NB * CC * HH_ * WW_) return;
  const int w = t % WW_;
  const int h = (t / WW_) % HH_;
  const int c = (t / (WW_ * HH_)) % CC;
  const int n = t / (WW_ * HH_ * CC);
  out[t] = X[((size_t)n * LL + (size_t)h * WW_ + w) * CC + c];
}

// ---------------------------------------------------------------------------
extern "C" void kernel_launch(void* const* d_in, const int* in_sizes, int n_in,
                              void* d_out, int out_size, void* d_ws, size_t ws_size,
                              hipStream_t stream) {
  (void)in_sizes; (void)n_in; (void)out_size; (void)ws_size;
  const float* feat0 = (const float*)d_in[0];
  const float* feat1 = (const float*)d_in[1];
  const float* Wq = (const float*)d_in[2];
  const float* Wk = (const float*)d_in[3];
  const float* Wv = (const float*)d_in[4];
  const float* Wm = (const float*)d_in[5];
  const float* W1 = (const float*)d_in[6];
  const float* W2 = (const float*)d_in[7];
  const float* g1 = (const float*)d_in[8];
  const float* b1 = (const float*)d_in[9];
  const float* g2 = (const float*)d_in[10];
  const float* b2 = (const float*)d_in[11];
  float* outp = (float*)d_out;

  const size_t NL = (size_t)NB * LL;
  size_t off = 0;
  char* base = (char*)d_ws;
  auto alloc = [&](size_t bytes) -> void* {
    void* p = base + off;
    off += (bytes + 255) & ~(size_t)255;
    return p;
  };
  float* Xf32   = (float*)alloc(NL * CC * 4);       // fp32 running f0
  bf16*  Concat = (bf16*) alloc(NL * 512 * 2);      // [x | msg] bf16
  bf16*  F1bf   = (bf16*) alloc(NL * CC * 2);
  bf16*  Qbf    = (bf16*) alloc(NL * CC * 2);       // [n,h,l,d]
  bf16*  Ktbf   = (bf16*) alloc(NL * CC * 2);       // [n,h,d,s]
  bf16*  Vbf    = (bf16*) alloc(NL * CC * 2);       // [n,h,s,d] (scaled 1/S)
  float* KVb    = (float*)alloc((size_t)NB * NHD * 1024 * 4);
  float* Ksum   = (float*)alloc((size_t)NB * NHD * 32 * 4);
  float* Zbuf   = (float*)alloc(NL * NHD * 4);
  bf16*  Attn   = (bf16*) alloc(NL * CC * 2);
  bf16*  FFN1   = (bf16*) alloc(NL * CC * 2);
  bf16*  WqP[2], *WkP[2], *WvP[2], *WmP[2], *W1P[2], *W2P[2];
  for (int i = 0; i < 2; ++i) {
    WqP[i] = (bf16*)alloc(65536 * 2);
    WkP[i] = (bf16*)alloc(65536 * 2);
    WvP[i] = (bf16*)alloc(65536 * 2);
    WmP[i] = (bf16*)alloc(65536 * 2);
    W1P[i] = (bf16*)alloc(131072 * 2);
    W2P[i] = (bf16*)alloc(65536 * 2);
  }

  // --- weight packing ---
  for (int i = 0; i < 2; ++i) {
    pack_w_kernel<<<256, 256, 0, stream>>>(Wq + (size_t)i * 65536, WqP[i], 256);
    pack_w_kernel<<<256, 256, 0, stream>>>(Wk + (size_t)i * 65536, WkP[i], 256);
    pack_w_kernel<<<256, 256, 0, stream>>>(Wv + (size_t)i * 65536, WvP[i], 256);
    pack_w_kernel<<<256, 256, 0, stream>>>(Wm + (size_t)i * 65536, WmP[i], 256);
    pack_w_kernel<<<512, 256, 0, stream>>>(W1 + (size_t)i * 131072, W1P[i], 512);
    pack_w_kernel<<<256, 256, 0, stream>>>(W2 + (size_t)i * 65536, W2P[i], 256);
  }

  // --- positional encoding + relayout ---
  const int tot = NB * CC * HH_ * WW_;
  pe_pack_kernel<<<(tot + 255) / 256, 256, 0, stream>>>(feat0, Xf32, Concat, 512, 1);
  pe_pack_kernel<<<(tot + 255) / 256, 256, 0, stream>>>(feat1, nullptr, F1bf, 256, 0);

  const dim3 gProj(2, (unsigned)(NL / 64));
  for (int i = 0; i < 2; ++i) {
    // q/k/v projections
    gemm_proj_kernel<<<gProj, 256, 0, stream>>>(Concat, 512, WqP[i], 256, 1, Qbf);
    gemm_proj_kernel<<<gProj, 256, 0, stream>>>(F1bf, 256, WkP[i], 256, 2, Ktbf);
    gemm_proj_kernel<<<gProj, 256, 0, stream>>>(F1bf, 256, WvP[i], 256, 3, Vbf);
    // linear attention core
    zero_kernel<<<(NB * NHD * 1024 + 255) / 256, 256, 0, stream>>>(KVb, NB * NHD * 1024);
    zero_kernel<<<(NB * NHD * 32 + 255) / 256, 256, 0, stream>>>(Ksum, NB * NHD * 32);
    kv_kernel<<<dim3(KV_SPLIT, NB * NHD), 128, 0, stream>>>(Ktbf, Vbf, KVb, Ksum);
    z_kernel<<<(NB * LL * NHD + 255) / 256, 256, 0, stream>>>(Qbf, Ksum, Zbuf);
    attn_kernel<<<dim3(LL / 64, NB * NHD), 128, 0, stream>>>(Qbf, KVb, Zbuf, Attn);
    // msg = LN1(attn @ Wm) -> Concat[:,256:]
    gemm_ln_kernel<<<(unsigned)(NL / 32), 256, 0, stream>>>(
        Attn, WmP[i], g1 + i * 256, b1 + i * 256, 0, Concat, nullptr);
    // FFN
    gemm_proj_kernel<<<gProj, 256, 0, stream>>>(Concat, 512, W1P[i], 512, 0, FFN1);
    gemm_ln_kernel<<<(unsigned)(NL / 32), 256, 0, stream>>>(
        FFN1, W2P[i], g2 + i * 256, b2 + i * 256, 1, Concat, Xf32);
  }

  final_kernel<<<(tot + 255) / 256, 256, 0, stream>>>(Xf32, outp);
}
